// LSTMPredictor_3917010174811
// MI455X (gfx1250) — compile-verified, matrix operand order checked by reading, not run on hardware
//
#include <hip/hip_runtime.h>

typedef __attribute__((ext_vector_type(16))) _Float16 v16h;
typedef __attribute__((ext_vector_type(8)))  float    v8f;

#define HID      150
#define G4       600
#define TSTEPS   1024
#define KP       160     // padded hidden (K) dim: 5 chunks of 32
#define NP       640     // padded 4H (N) dim: 40 tiles of 16
#define MROWS    16      // batch rows per workgroup
#define NWG      16      // 16 workgroups * 16 rows = 256 batch
#define NTHREADS 512     // 16 waves -> 4 waves/SIMD

__device__ __forceinline__ float sigmoidf_(float x) {
    return 1.0f / (1.0f + __expf(-x));
}

// gates += h * W^T for NT register-resident N tiles; A double-buffered from LDS
template<int NT>
__device__ __forceinline__ void gemm_step(const _Float16* __restrict__ aBase,
                                          const v16h (&breg)[3][5], v8f (&acc)[3]) {
    v16h a_nxt = *reinterpret_cast<const v16h*>(aBase);
    #pragma unroll
    for (int kc = 0; kc < 5; ++kc) {
        const v16h a_cur = a_nxt;
        if (kc < 4)                                     // next A chunk flies under WMMAs
            a_nxt = *reinterpret_cast<const v16h*>(aBase + (kc + 1) * 32);
        #pragma unroll
        for (int j = 0; j < NT; ++j)
            acc[j] = __builtin_amdgcn_wmma_f32_16x16x32_f16(
                false, a_cur, false, breg[j][kc], (short)0, acc[j], false, false);
    }
}

__global__ __launch_bounds__(NTHREADS, 1)
void lstm_persistent(const float* __restrict__ input,   // [256,1024]
                     const float* __restrict__ W_ih,    // [600,1]
                     const float* __restrict__ W_hh,    // [600,150]
                     const float* __restrict__ b_ih,    // [600]
                     const float* __restrict__ b_hh,    // [600]
                     const float* __restrict__ W_lin,   // [1,150]
                     const float* __restrict__ b_lin,   // [1]
                     float* __restrict__ out)           // [256,1024]
{
    // 62.2 KB static LDS (W_hh lives in registers, not LDS)
    __shared__ float                  lds_gate[MROWS * NP];   // gate pre-activations
    __shared__ float                  lds_c[MROWS * KP];      // cell state
    __shared__ __align__(32) _Float16 lds_h[MROWS * KP];      // hidden state (A operand)
    __shared__ float                  lds_wih[NP];
    __shared__ float                  lds_bias[NP];
    __shared__ float                  lds_wlin[KP];
    __shared__ float                  lds_x[MROWS];
    __shared__ float                  lds_out[MROWS];

    const int tid  = threadIdx.x;
    const int lane = tid & 31;
    const int wave = tid >> 5;           // 0..15
    const int b0   = blockIdx.x * MROWS;
    const float blin = b_lin[0];

    const int bcol = lane & 15;          // B/A: lane -> column / row within tile
    const int half = lane >> 4;          // upper lanes take second k group

    // scalar (SGPR) wave id -> provably uniform branch, keeps EXEC all-ones at WMMAs
    const int  wave_s = __builtin_amdgcn_readfirstlane(tid) >> 5;
    const bool has3   = (wave_s < 8);    // waves 0-7 own a third N tile

    // ---- one-time init: small tables + state ----
    for (int idx = tid; idx < NP; idx += NTHREADS) {
        lds_wih[idx]  = (idx < G4) ? W_ih[idx] : 0.0f;
        lds_bias[idx] = (idx < G4) ? (b_ih[idx] + b_hh[idx]) : 0.0f;
    }
    for (int idx = tid; idx < KP; idx += NTHREADS)
        lds_wlin[idx] = (idx < HID) ? W_lin[idx] : 0.0f;
    for (int idx = tid; idx < MROWS * KP; idx += NTHREADS) {
        lds_h[idx] = (_Float16)0.0f;     // h0 = 0 (k padding stays 0 forever)
        lds_c[idx] = 0.0f;               // c0 = 0
    }

    // ---- one-time init: W_hh fp32 -> f16, persistent in VGPRs ----
    // wave w owns N tiles {w, w+16} and (w<8) {w+32}; lane -> column n of tile
    v16h breg[3][5];
    #pragma unroll
    for (int j = 0; j < 3; ++j) {
        const int nt = wave + j * 16;
        const int n  = nt * 16 + bcol;
        #pragma unroll
        for (int kc = 0; kc < 5; ++kc) {
            v16h bm;
            #pragma unroll
            for (int e = 0; e < 16; ++e) {
                const int k = kc * 32 + half * 16 + e;
                const float v = (nt < 40 && n < G4 && k < HID) ? W_hh[n * HID + k] : 0.0f;
                bm[e] = (_Float16)v;
            }
            breg[j][kc] = bm;
        }
    }

    const _Float16* aBase = &lds_h[bcol * KP + half * 16];

    for (int t = 0; t < TSTEPS; ++t) {
        // barrier A: previous step's h/c writes and lds_out partials complete
        __syncthreads();

        if (tid < MROWS) {
            if (t > 0)                                   // drain step t-1 (overlaps matmul)
                out[(b0 + tid) * TSTEPS + (t - 1)] = lds_out[tid] + blin;
            lds_out[tid] = 0.0f;
            lds_x[tid]   = input[(b0 + tid) * TSTEPS + t];
        }
        if (t + 8 < TSTEPS)   // stream the scalar input ahead
            __builtin_prefetch(&input[(b0 + (tid & 15)) * TSTEPS + t + 8], 0, 1);

        // ---- recurrent GEMM: one scalar branch per step, branch-free inner loops ----
        v8f acc[3] = { v8f{}, v8f{}, v8f{} };
        if (has3) gemm_step<3>(aBase, breg, acc);
        else      gemm_step<2>(aBase, breg, acc);

        // spill gate pre-activations (C layout: lane->col, vgpr->row)
        {
            const int rbase = half * 8;
            const int c0 = (wave +  0) * 16 + bcol;
            const int c1 = (wave + 16) * 16 + bcol;
            const int c2 = (wave + 32) * 16 + bcol;
            #pragma unroll
            for (int v = 0; v < 8; ++v) {
                lds_gate[(rbase + v) * NP + c0] = acc[0][v];
                lds_gate[(rbase + v) * NP + c1] = acc[1][v];
            }
            if (has3) {
                #pragma unroll
                for (int v = 0; v < 8; ++v)
                    lds_gate[(rbase + v) * NP + c2] = acc[2][v];
            }
        }

        // barrier B: gates + x visible
        __syncthreads();

        // ---- elementwise LSTM cell + projection: 32 threads per batch row ----
        {
            const int r  = tid >> 5;
            const int nl = tid & 31;
            const float x = lds_x[r];
            float part = 0.0f;
            for (int n = nl; n < HID; n += 32) {
                const float gi = lds_gate[r * NP + n          ] + x * lds_wih[n          ] + lds_bias[n          ];
                const float gf = lds_gate[r * NP + n +     HID] + x * lds_wih[n +     HID] + lds_bias[n +     HID];
                const float gg = lds_gate[r * NP + n + 2 * HID] + x * lds_wih[n + 2 * HID] + lds_bias[n + 2 * HID];
                const float go = lds_gate[r * NP + n + 3 * HID] + x * lds_wih[n + 3 * HID] + lds_bias[n + 3 * HID];
                const float c  = sigmoidf_(gf) * lds_c[r * KP + n] + sigmoidf_(gi) * tanhf(gg);
                lds_c[r * KP + n] = c;
                const float h  = sigmoidf_(go) * tanhf(c);
                lds_h[r * KP + n] = (_Float16)h;       // next step's A operand
                part += h * lds_wlin[n];               // output projection partial
            }
            atomicAdd(&lds_out[r], part);              // ds_add_f32 reduction
        }
    }

    // drain final step
    __syncthreads();
    if (tid < MROWS)
        out[(b0 + tid) * TSTEPS + (TSTEPS - 1)] = lds_out[tid] + blin;
}

extern "C" void kernel_launch(void* const* d_in, const int* in_sizes, int n_in,
                              void* d_out, int out_size, void* d_ws, size_t ws_size,
                              hipStream_t stream) {
    const float* input = (const float*)d_in[0];
    const float* W_ih  = (const float*)d_in[1];
    const float* W_hh  = (const float*)d_in[2];
    const float* b_ih  = (const float*)d_in[3];
    const float* b_hh  = (const float*)d_in[4];
    const float* W_lin = (const float*)d_in[5];
    const float* b_lin = (const float*)d_in[6];
    float* out = (float*)d_out;

    lstm_persistent<<<dim3(NWG), dim3(NTHREADS), 0, stream>>>(
        input, W_ih, W_hh, b_ih, b_hh, W_lin, b_lin, out);
}
